// calculate_precision_7593502179517
// MI455X (gfx1250) — matrix operand order repaired and checked by reading, not compiled, not run on hardware
//
#include <hip/hip_runtime.h>

typedef __attribute__((ext_vector_type(16))) _Float16 v16h;
typedef __attribute__((ext_vector_type(8)))  float    v8f;

#define N_COLS    26   // physical row stride of both inputs (second dim)
#define EVAL_ROWS 32   // reference evaluates only the first 32 rows

// Single wave32, single workgroup. Computes per-class tp1/pp1/s1 over the
// 32x26 live region with V_WMMA_F32_16X16X32_F16 (K=32 == EVAL_ROWS, 0/1
// values exact in f16, f32 accumulate), then finishes the weighted-precision
// formula on lanes 0..25.
__global__ __launch_bounds__(32)
void weighted_precision_wmma(const float* __restrict__ scores,
                             const int*   __restrict__ labels,
                             const int*   __restrict__ ncp,
                             float*       __restrict__ out) {
  const int L     = threadIdx.x;          // lane id, wave32
  const int nc    = ncp[0];               // n_classes (26)
  const int nwin  = L & 15;               // column / class-within-group
  const int rbase = (L < 16) ? 0 : 16;    // B-matrix: K half held by this lane
  const int aoff  = (L < 16) ? 0 : 8;     // unused; A also takes its K half from rbase

  __shared__ float s_tp[32];
  __shared__ float s_pp[32];
  __shared__ float s_s1[32];

  const _Float16 h1 = (_Float16)1.0f;
  const _Float16 h0 = (_Float16)0.0f;

  v16h aOnes;
  #pragma unroll
  for (int e = 0; e < 16; ++e) aOnes[e] = h1;

  #pragma unroll
  for (int g = 0; g < 2; ++g) {           // class groups 0..15, 16..31(pad)
    const int  c      = 16 * g + nwin;
    const bool cvalid = (c < nc);
    const int  cc     = cvalid ? c : 0;   // clamp address, zero value -> no
                                          // EXEC divergence before WMMA

    // B matrices (32x16 f16, K=row, N=class): lane holds column nwin,
    // rows rbase..rbase+15; element e packs into VGPR e/2, half e%2.
    v16h bP, bY;
    #pragma unroll
    for (int e = 0; e < 16; ++e) {
      const int   r = rbase + e;
      const float f = scores[r * N_COLS + cc];
      const int   y = labels[r * N_COLS + cc];
      bP[e] = (cvalid && (f >= 0.5f)) ? h1 : h0;
      bY[e] = (cvalid && (y != 0))    ? h1 : h0;
    }

    // A = Y^T (16x32 f16, M=class, K=row): lane holds row M=nwin,
    // K = rbase+e (same contiguous K order as B).
    v16h aY;
    #pragma unroll
    for (int e = 0; e < 16; ++e) {
      const int k = rbase + e;
      const int y = labels[k * N_COLS + cc];
      aY[e] = (cvalid && (y != 0)) ? h1 : h0;
    }

    v8f zacc = {};
    // diag(d_tp) = tp1 ; every row of d_pp = pp1 ; every row of d_s = s1
    v8f d_tp = __builtin_amdgcn_wmma_f32_16x16x32_f16(false, aY,    false, bP,
                                                      (short)0, zacc, false, false);
    v8f d_pp = __builtin_amdgcn_wmma_f32_16x16x32_f16(false, aOnes, false, bP,
                                                      (short)0, zacc, false, false);
    v8f d_s  = __builtin_amdgcn_wmma_f32_16x16x32_f16(false, aOnes, false, bY,
                                                      (short)0, zacc, false, false);

    // D layout: element v of lane L is D[m][n], m = v + (L<16?0:8), n = nwin.
    if (L < 8)        s_tp[16 * g + L]        = d_tp[L];       // m = v = n = L
    else if (L >= 24) s_tp[16 * g + (L - 16)] = d_tp[L - 24];  // m = v+8 = n = L-16
    if (L < 16) {
      s_pp[16 * g + L] = d_pp[0];   // any row replicates the column sum
      s_s1[16 * g + L] = d_s[0];
    }
  }

  __syncthreads();

  if (L < nc) {
    const float n   = (float)EVAL_ROWS;
    const float tp1 = s_tp[L];
    const float pp1 = s_pp[L];
    const float s1  = s_s1[L];
    const float tp0 = n - s1 - pp1 + tp1;   // sum (1-y)(1-p)
    const float pp0 = n - pp1;
    const float s0  = n - s1;
    const float p1  = (pp1 > 0.0f) ? (tp1 / pp1) : 0.0f;
    const float p0  = (pp0 > 0.0f) ? (tp0 / pp0) : 0.0f;
    out[L] = (s1 * p1 + s0 * p0) / n;
  }
}

extern "C" void kernel_launch(void* const* d_in, const int* in_sizes, int n_in,
                              void* d_out, int out_size, void* d_ws, size_t ws_size,
                              hipStream_t stream) {
  (void)in_sizes; (void)n_in; (void)out_size; (void)d_ws; (void)ws_size;
  const float* scores = (const float*)d_in[0];
  const int*   labels = (const int*)d_in[1];
  const int*   ncp    = (const int*)d_in[2];
  float*       out    = (float*)d_out;

  // One wave32 on one WGP: the live working set is ~6.7 KB, so this is a
  // launch-latency-bound kernel; 6 WMMAs replace all 3x26 column reductions.
  weighted_precision_wmma<<<1, 32, 0, stream>>>(scores, labels, ncp, out);
}
